// MentionScore_74036646249298
// MI455X (gfx1250) — compile-verified, matrix operand order checked by reading, not run on hardware
//
#include <hip/hip_runtime.h>

// ---------------------------------------------------------------------------
// MentionScore for MI455X (gfx1250, wave32, WMMA bf16 16x16x32 / f32 acc)
//
//   0) weight prep: swizzle aW1/aW2/mW1/mW2 into bf16 WMMA-frag layout (once)
//   1) h1a = relu(states @ aW1 + ab1)           [8192,512]x[512,150]   (WMMA)
//   2) h2a = relu(h1a    @ aW2 + ab2)           [8192,160]x[150,150]   (WMMA)
//   3) attns = h2a @ aW3 + ab3                  [8192,150]x[150,1]     (dot)
//   4) g_i  = span gather / softmax / pooled embeds / dist  -> d_out   (wave/span)
//   5) Ym   = relu(g_i @ mW1 + mb1)             [65536,1344]x[1344,150] (WMMA)
//   6) Zm   = relu(Ym  @ mW2 + mb2)             [65536,160]x[150,150]   (WMMA)
//   7) scores = Zm @ mW3 + mb3 -> d_out tail                            (dot)
//
// GEMM: no LDS, no barriers. B fragments are pre-swizzled bf16 in global
// (L2-resident), loaded straight to VGPRs (2 x b128 per frag, lane-coalesced).
// A is f32 in global, converted to bf16 in VALU (co-executes with WMMA).
// ---------------------------------------------------------------------------

typedef __attribute__((ext_vector_type(16))) __bf16          bf16x16;
typedef __attribute__((ext_vector_type(16))) unsigned short  u16x16;
typedef __attribute__((ext_vector_type(8)))  float           f32x8;
typedef __attribute__((ext_vector_type(4)))  float           f32x4;

__device__ __forceinline__ unsigned short f2bf(float f) {
    unsigned int u = __builtin_bit_cast(unsigned int, f);
    u += 0x7FFFu + ((u >> 16) & 1u);            // round-to-nearest-even
    return (unsigned short)(u >> 16);
}

#define NTILE 10     // N padded to 160 = 10 x 16
#define MTILE 128    // 8 waves x 16 rows

// Pre-swizzle W[kvalid x nvalid] (f32, row-major) into bf16 WMMA B-fragment
// layout: out[((kt*NTILE + n)*32 + lane)*16 + i], zero-padded to Kpad x 160.
__global__ __launch_bounds__(256)
void swizzle_w_kernel(const float* __restrict__ W, int kvalid, int nvalid,
                      int nKt, unsigned short* __restrict__ out)
{
    const int e = blockIdx.x * 256 + threadIdx.x;
    const int total = nKt * NTILE * 32 * 16;
    if (e >= total) return;
    const int i    = e & 15;
    const int lane = (e >> 4) & 31;
    const int q    = e >> 9;            // kt*NTILE + n
    const int n    = q % NTILE;
    const int kt   = q / NTILE;
    const int lg   = lane >> 4;
    const int kk   = (i & 7) + lg * 8 + ((i >> 3) << 4);   // K within 32-tile
    const int kg   = kt * 32 + kk;
    const int col  = n * 16 + (lane & 15);
    const float v  = (kg < kvalid && col < nvalid) ? W[kg * nvalid + col] : 0.0f;
    out[e] = f2bf(v);
}

// out = relu(A[M x Kpad](f32) @ Bsw + bias), out ld=160 f32,
// columns nvalid..159 written as zero (next layer uses Kpad=160 unmasked).
__global__ __launch_bounds__(256)
void gemm_bias_relu_kernel(const float* __restrict__ A, int lda, int Kpad,
                           const unsigned short* __restrict__ Bsw, int nvalid,
                           const float* __restrict__ bias,
                           float* __restrict__ out)
{
    const int lane    = threadIdx.x & 31;
    const int wave    = threadIdx.x >> 5;
    const int laneHi  = lane >> 4;                 // 0 or 1
    const int rowBase = blockIdx.x * MTILE + wave * 16;
    const int mrow    = rowBase + (lane & 15);

    f32x8 acc[NTILE] = {};
    const int nKt = Kpad >> 5;
    const float* arow0 = A + (size_t)mrow * lda + laneHi * 8;

    for (int kt = 0; kt < nKt; ++kt) {
        // ---- A fragment: f32 global -> bf16 regs
        const float* arow = arow0 + kt * 32;
        const f32x4 a0 = *(const f32x4*)(arow + 0);
        const f32x4 a1 = *(const f32x4*)(arow + 4);
        const f32x4 a2 = *(const f32x4*)(arow + 16);
        const f32x4 a3 = *(const f32x4*)(arow + 20);
        u16x16 au;
        #pragma unroll
        for (int j = 0; j < 4; ++j) {
            au[j]      = f2bf(a0[j]);
            au[4 + j]  = f2bf(a1[j]);
            au[8 + j]  = f2bf(a2[j]);
            au[12 + j] = f2bf(a3[j]);
        }
        const bf16x16 afrag = __builtin_bit_cast(bf16x16, au);

        // ---- all 10 B fragments straight from global (L2-resident),
        //      then a back-to-back WMMA accumulator chain
        const unsigned short* bbase =
            Bsw + ((size_t)kt * NTILE * 32 + lane) * 16;
        u16x16 b[NTILE];
        #pragma unroll
        for (int n = 0; n < NTILE; ++n)
            b[n] = *(const u16x16*)(bbase + n * 32 * 16);

        #pragma unroll
        for (int n = 0; n < NTILE; ++n) {
            const bf16x16 bfrag = __builtin_bit_cast(bf16x16, b[n]);
            acc[n] = __builtin_amdgcn_wmma_f32_16x16x32_bf16(
                         false, afrag, false, bfrag, (short)0, acc[n],
                         false, false);
        }
    }

    // ---- epilogue: bias + relu; zero the padded columns
    const int colLo  = lane & 15;
    const int rowOut = rowBase + laneHi * 8;
    #pragma unroll
    for (int n = 0; n < NTILE; ++n) {
        const int col = n * 16 + colLo;
        const bool ok = (col < nvalid);
        const float bv = ok ? bias[col] : 0.0f;
        #pragma unroll
        for (int r = 0; r < 8; ++r) {
            float v = acc[n][r] + bv;
            v = v > 0.0f ? v : 0.0f;
            out[(size_t)(rowOut + r) * 160 + col] = ok ? v : 0.0f;
        }
    }
}

// one wave per span: build g_i = [states[start] | states[end] | attnEmb | dist]
__global__ __launch_bounds__(256)
void span_kernel(const float* __restrict__ states, const float* __restrict__ embeds,
                 const float* __restrict__ attns, const float* __restrict__ dist_table,
                 const int* __restrict__ starts, const int* __restrict__ offs,
                 float* __restrict__ g_i)
{
    const int s    = (blockIdx.x * blockDim.x + threadIdx.x) >> 5;
    const int lane = threadIdx.x & 31;
    if (s >= 65536) return;

    const int start = starts[s];
    const int off   = offs[s];
    const int len   = off + 1;
    const int end   = min(start + off, 8191);

    // width-8 masked softmax (uniform per wave; redundant per lane)
    float w[8];
    float mx = -1e30f;
    #pragma unroll
    for (int j = 0; j < 8; ++j) {
        const int idx = min(start + j, 8191);
        const float v = (j < len) ? attns[idx] : -1e10f;
        w[j] = v;
        mx = fmaxf(mx, v);
    }
    float sum = 0.0f;
    #pragma unroll
    for (int j = 0; j < 8; ++j) { w[j] = __expf(w[j] - mx); sum += w[j]; }
    const float inv = 1.0f / sum;
    #pragma unroll
    for (int j = 0; j < 8; ++j) w[j] *= inv;

    float* g = g_i + (size_t)s * 1344;

    // copy start/end LSTM states (512 floats each), float4-vectorized
    const f32x4* ps = (const f32x4*)(states + (size_t)start * 512);
    const f32x4* pe = (const f32x4*)(states + (size_t)end   * 512);
    f32x4* g0 = (f32x4*)g;
    f32x4* g1 = (f32x4*)(g + 512);
    #pragma unroll
    for (int j = 0; j < 4; ++j) {
        g0[lane + 32 * j] = ps[lane + 32 * j];
        g1[lane + 32 * j] = pe[lane + 32 * j];
    }

    // attention-weighted embedding pool (300 cols)
    for (int c = lane; c < 300; c += 32) {
        float a = 0.0f;
        #pragma unroll
        for (int j = 0; j < 8; ++j) {
            if (j < len) {
                const int idx = min(start + j, 8191);
                a += w[j] * embeds[(size_t)idx * 300 + c];
            }
        }
        g[1024 + c] = a;
    }

    // distance feature: bin = #bins <= len ; BINS = 1,2,3,4,5,8,16,32,64
    const int bin = (len >= 1) + (len >= 2) + (len >= 3) + (len >= 4) + (len >= 5) +
                    (len >= 8) + (len >= 16) + (len >= 32) + (len >= 64);
    if (lane < 20) g[1324 + lane] = dist_table[bin * 20 + lane];
}

// final 150 -> 1 projection (no relu)
__global__ __launch_bounds__(256)
void dot_kernel(const float* __restrict__ A, int lda, int kvalid,
                const float* __restrict__ w, const float* __restrict__ b,
                float* __restrict__ out, int M)
{
    const int r = blockIdx.x * blockDim.x + threadIdx.x;
    if (r >= M) return;
    const float* a = A + (size_t)r * lda;
    float acc = 0.0f;
    for (int k = 0; k < kvalid; ++k) acc += a[k] * w[k];
    out[r] = acc + b[0];
}

extern "C" void kernel_launch(void* const* d_in, const int* in_sizes, int n_in,
                              void* d_out, int out_size, void* d_ws, size_t ws_size,
                              hipStream_t stream) {
    const float* states = (const float*)d_in[0];
    const float* embeds = (const float*)d_in[1];
    const float* aW1 = (const float*)d_in[2];  const float* ab1 = (const float*)d_in[3];
    const float* aW2 = (const float*)d_in[4];  const float* ab2 = (const float*)d_in[5];
    const float* aW3 = (const float*)d_in[6];  const float* ab3 = (const float*)d_in[7];
    const float* mW1 = (const float*)d_in[8];  const float* mb1 = (const float*)d_in[9];
    const float* mW2 = (const float*)d_in[10]; const float* mb2 = (const float*)d_in[11];
    const float* mW3 = (const float*)d_in[12]; const float* mb3 = (const float*)d_in[13];
    const float* dist = (const float*)d_in[14];
    const int* sst = (const int*)d_in[15];
    const int* sof = (const int*)d_in[16];
    float* out = (float*)d_out;

    // workspace layout (float units; all offsets 32B-aligned)
    float* ws    = (float*)d_ws;
    float* h1a   = ws;                                 // 8192*160
    float* h2a   = h1a + (size_t)8192 * 160;           // 8192*160
    float* attns = h2a + (size_t)8192 * 160;           // 8192
    float* Ym    = attns + 8192;                       // 65536*160
    float* Zm    = Ym + (size_t)65536 * 160;           // 65536*160
    unsigned short* usw = (unsigned short*)(Zm + (size_t)65536 * 160);
    unsigned short* aW1sw = usw;                       // 16*5120 bf16
    unsigned short* aW2sw = aW1sw + 16 * 5120;         //  5*5120
    unsigned short* mW1sw = aW2sw + 5 * 5120;          // 42*5120
    unsigned short* mW2sw = mW1sw + 42 * 5120;         //  5*5120

    const dim3 blk(256);

    // weight prep (tiny, runs once per call)
    swizzle_w_kernel<<<(16 * 5120 + 255) / 256, blk, 0, stream>>>(aW1, 512, 150, 16, aW1sw);
    swizzle_w_kernel<<<( 5 * 5120 + 255) / 256, blk, 0, stream>>>(aW2, 150, 150,  5, aW2sw);
    swizzle_w_kernel<<<(42 * 5120 + 255) / 256, blk, 0, stream>>>(mW1, 1344, 150, 42, mW1sw);
    swizzle_w_kernel<<<( 5 * 5120 + 255) / 256, blk, 0, stream>>>(mW2, 150, 150,  5, mW2sw);

    // attention MLP
    gemm_bias_relu_kernel<<<8192 / MTILE, blk, 0, stream>>>(states, 512, 512,
                                                            aW1sw, 150, ab1, h1a);
    gemm_bias_relu_kernel<<<8192 / MTILE, blk, 0, stream>>>(h1a, 160, 160,
                                                            aW2sw, 150, ab2, h2a);
    dot_kernel<<<8192 / 256, blk, 0, stream>>>(h2a, 160, 150, aW3, ab3, attns, 8192);

    // span features -> g_i (first output)
    span_kernel<<<(65536 * 32) / 256, blk, 0, stream>>>(states, embeds, attns, dist,
                                                        sst, sof, out);

    // mention MLP over g_i
    gemm_bias_relu_kernel<<<65536 / MTILE, blk, 0, stream>>>(out, 1344, 1344,
                                                             mW1sw, 150, mb1, Ym);
    gemm_bias_relu_kernel<<<65536 / MTILE, blk, 0, stream>>>(Ym, 160, 160,
                                                             mW2sw, 150, mb2, Zm);
    dot_kernel<<<65536 / 256, blk, 0, stream>>>(Zm, 160, 150, mW3, mb3,
                                                out + (size_t)65536 * 1344, 65536);
}
